// MinkUNet_35820027248902
// MI455X (gfx1250) — matrix operand order repaired and checked by reading, not compiled, not run on hardware
//
#include <hip/hip_runtime.h>

// ---------------------------------------------------------------------------
// MinkUNet (dense-grid masked 3D UNet) for gfx1250 (CDNA5, wave32, WMMA).
// Implicit-GEMM convs on v_wmma_f32_16x16x32_bf16:
//   - f32 activations/weights in HBM (net is HBM-bound at outer levels),
//   - branch-free predication (clamped address + select-multiply),
//   - vectorized float4 A loads; B slab staged in LDS in fragment order
//     (4 waves/block share it; double-buffered, one barrier per K-chunk),
//   - 2 M-tiles per wave: two accumulators share each B fragment ->
//     back-to-back independent WMMAs, half the barriers/weight traffic.
// Masked BN uses LDS tree reductions + global float atomics.
// ---------------------------------------------------------------------------

typedef __attribute__((ext_vector_type(16))) __bf16 v16bf;
typedef __attribute__((ext_vector_type(8)))  float  v8f;

#define CMAX 384  // max channel count (u1 concat: 256+128)

__device__ __forceinline__ v8f wmma_bf16(v16bf a, v16bf b, v8f c) {
  // (neg_a, A, neg_b, B, c_mod, C, reuse_a, reuse_b)
  return __builtin_amdgcn_wmma_f32_16x16x32_bf16(false, a, false, b, (short)0, c,
                                                 false, false);
}

// ---------------------------------------------------------------------------
// CDNA5 WMMA fragment layouts (ISA 7.12.2, wave32):
//  A (16x32 bf16): lane l holds row M=l&15; element i maps to
//    K(l,i) = (i&7) + ((i>>3)<<4) + ((l>>4)<<3)
//  B (32x16 bf16): lane l holds col N=l&15; K(l,i) = (i&7) + ((i>>3)<<4) + ((l>>4)<<3)
//  C/D (16x16 f32): lane l col N=l&15; vgpr r row M = r + ((l>>4)<<3)
// ---------------------------------------------------------------------------

// Vectorized A fragment: Ci % 32 == 0, base ka is 8-aligned -> float4 loads.
__device__ __forceinline__ v16bf load_A_vec(const float* __restrict__ ap, int ka,
                                            float asel) {
  const float4* p0 = (const float4*)(ap + ka);
  const float4* p1 = (const float4*)(ap + ka + 16);
  float4 a0 = p0[0], a1 = p0[1];
  float4 a2 = p1[0], a3 = p1[1];
  v16bf A;
  A[0]  = (__bf16)(a0.x * asel); A[1]  = (__bf16)(a0.y * asel);
  A[2]  = (__bf16)(a0.z * asel); A[3]  = (__bf16)(a0.w * asel);
  A[4]  = (__bf16)(a1.x * asel); A[5]  = (__bf16)(a1.y * asel);
  A[6]  = (__bf16)(a1.z * asel); A[7]  = (__bf16)(a1.w * asel);
  A[8]  = (__bf16)(a2.x * asel); A[9]  = (__bf16)(a2.y * asel);
  A[10] = (__bf16)(a2.z * asel); A[11] = (__bf16)(a2.w * asel);
  A[12] = (__bf16)(a3.x * asel); A[13] = (__bf16)(a3.y * asel);
  A[14] = (__bf16)(a3.z * asel); A[15] = (__bf16)(a3.w * asel);
  return A;
}

// Guarded A fragment (stem, Ci=5): clamped index + select-multiply, branch-free.
__device__ __forceinline__ v16bf load_A_guard(const float* __restrict__ ap, int ka,
                                              int Ci, float asel) {
  v16bf A;
#pragma unroll
  for (int j = 0; j < 8; ++j) {
    int k1 = ka + j, k2 = ka + 16 + j;
    float f1 = ap[k1 < Ci ? k1 : 0] * ((k1 < Ci) ? asel : 0.0f);
    float f2 = ap[k2 < Ci ? k2 : 0] * ((k2 < Ci) ? asel : 0.0f);
    A[j] = (__bf16)f1;
    A[8 + j] = (__bf16)f2;
  }
  return A;
}

// Block-cooperative B slab stage: 32(K) x 16(N) floats -> bf16 LDS in the
// per-lane fragment order, so each lane later reads one contiguous v16bf.
// 128 threads; each thread fetches 4 contiguous columns for one K value
// (single float4 in the fast path; clamped loads * 0/1 selector at edges --
// loads stay unconditional, no exec-mask predication).
__device__ __forceinline__ void stage_B(const float* __restrict__ wp, int k0,
                                        int Ci, int Co, int col0,
                                        __bf16* __restrict__ buf, int tid) {
  int k  = tid >> 2;          // 0..31
  int n0 = (tid & 3) * 4;     // 0,4,8,12
  int kk = k0 + k;
  int cb = col0 + n0;
  int lane = n0 + (((k >> 3) & 1) << 4);  // B layout: lanes 16-31 hold K 8..15,24..31
  int i = (k & 7) + ((k >> 4) << 3);
  float f0, f1, f2, f3;
  if (((Co & 3) == 0) && (kk < Ci) && (cb + 3 < Co)) {
    float4 w4 = *(const float4*)(wp + (size_t)kk * Co + cb);
    f0 = w4.x; f1 = w4.y; f2 = w4.z; f3 = w4.w;
  } else {
    int kc = kk < Ci ? kk : 0;
    const float* p = wp + (size_t)kc * Co;
    float ksel = (kk < Ci) ? 1.0f : 0.0f;
    float s0 = (cb + 0 < Co) ? ksel : 0.0f;
    float s1 = (cb + 1 < Co) ? ksel : 0.0f;
    float s2 = (cb + 2 < Co) ? ksel : 0.0f;
    float s3 = (cb + 3 < Co) ? ksel : 0.0f;
    float t0v = p[cb + 0 < Co ? cb + 0 : 0];
    float t1v = p[cb + 1 < Co ? cb + 1 : 0];
    float t2v = p[cb + 2 < Co ? cb + 2 : 0];
    float t3v = p[cb + 3 < Co ? cb + 3 : 0];
    f0 = t0v * s0; f1 = t1v * s1; f2 = t2v * s2; f3 = t3v * s3;
  }
  buf[(lane + 0) * 16 + i] = (__bf16)f0;
  buf[(lane + 1) * 16 + i] = (__bf16)f1;
  buf[(lane + 2) * 16 + i] = (__bf16)f2;
  buf[(lane + 3) * 16 + i] = (__bf16)f3;
}

__device__ __forceinline__ v16bf read_B(const __bf16* __restrict__ buf, int lane) {
  return *(const v16bf*)(buf + lane * 16);
}

// 3x3x3 SAME conv, stride 1. grid = (ceil(nv/128), ceil(Co/16)), block = 128.
__global__ void __launch_bounds__(128) conv3s_wmma(
    const float* __restrict__ in, const float* __restrict__ w,
    float* __restrict__ out, int D, int Ci, int Co, int os, int nv) {
  __shared__ alignas(32) __bf16 sB[2][512];
  const int tid  = threadIdx.x;
  const int lane = tid & 31;
  const int wave = tid >> 5;
  const int row  = lane & 15;
  const int half = lane >> 4;
  const int col0 = blockIdx.y << 4;
  const int col  = col0 + row;
  const int vbase = ((blockIdx.x * 4 + wave) * 2) << 4;  // 2 tiles of 16 voxels
  const int HW = D * D;
  const bool kguard = (Ci & 31) != 0;

  int zc[2], yc[2], xc[2];
  float rok[2];
#pragma unroll
  for (int tl = 0; tl < 2; ++tl) {
    int vv = vbase + tl * 16 + row;
    rok[tl] = (vv < nv) ? 1.0f : 0.0f;
    if (vv >= nv) vv = nv - 1;
    int zz = vv / HW; int rem = vv - zz * HW;
    zc[tl] = zz; yc[tl] = rem / D; xc[tl] = rem - (rem / D) * D;
  }

  v8f acc0 = {}, acc1 = {};
  int it = 0;
  for (int t = 0; t < 27; ++t) {
    int dz = t / 9 - 1, dy = (t / 3) % 3 - 1, dx = t % 3 - 1;
    const float* ap[2];
    float asel[2];
#pragma unroll
    for (int tl = 0; tl < 2; ++tl) {
      int zz = zc[tl] + dz, yv = yc[tl] + dy, xv = xc[tl] + dx;
      bool inb = (unsigned)zz < (unsigned)D && (unsigned)yv < (unsigned)D &&
                 (unsigned)xv < (unsigned)D;
      asel[tl] = inb ? rok[tl] : 0.0f;
      size_t vox = inb ? (size_t)((zz * D + yv) * D + xv) : 0;
      ap[tl] = in + vox * Ci;
    }
    const float* wp = w + (size_t)t * Ci * Co;
    for (int k0 = 0; k0 < Ci; k0 += 32, ++it) {
      __bf16* buf = &sB[it & 1][0];
      stage_B(wp, k0, Ci, Co, col0, buf, tid);
      __syncthreads();
      int ka = k0 + half * 8;
      v16bf B  = read_B(buf, lane);
      v16bf A0 = kguard ? load_A_guard(ap[0], ka, Ci, asel[0])
                        : load_A_vec(ap[0], ka, asel[0]);
      v16bf A1 = kguard ? load_A_guard(ap[1], ka, Ci, asel[1])
                        : load_A_vec(ap[1], ka, asel[1]);
      acc0 = wmma_bf16(A0, B, acc0);
      acc1 = wmma_bf16(A1, B, acc1);
    }
  }
  if (col < Co) {
#pragma unroll
    for (int r = 0; r < 8; ++r) {
      int m = r + half * 8;
      int v0 = vbase + m, v1 = vbase + 16 + m;
      if (v0 < nv) out[(size_t)v0 * os + col] = acc0[r];
      if (v1 < nv) out[(size_t)v1 * os + col] = acc1[r];
    }
  }
}

// 2x2x2 stride-2 conv (VALID). grid over output voxels.
__global__ void __launch_bounds__(128) conv2d_wmma(
    const float* __restrict__ in, const float* __restrict__ w,
    float* __restrict__ out, int Din, int Ci, int Co, int os, int nv) {
  __shared__ alignas(32) __bf16 sB[2][512];
  const int Do = Din >> 1;
  const int tid  = threadIdx.x;
  const int lane = tid & 31;
  const int wave = tid >> 5;
  const int row  = lane & 15;
  const int half = lane >> 4;
  const int col0 = blockIdx.y << 4;
  const int col  = col0 + row;
  const int vbase = ((blockIdx.x * 4 + wave) * 2) << 4;
  const int HW = Do * Do;

  int zc[2], yc[2], xc[2];
  float rok[2];
#pragma unroll
  for (int tl = 0; tl < 2; ++tl) {
    int vv = vbase + tl * 16 + row;
    rok[tl] = (vv < nv) ? 1.0f : 0.0f;
    if (vv >= nv) vv = nv - 1;
    int zz = vv / HW; int rem = vv - zz * HW;
    zc[tl] = zz; yc[tl] = rem / Do; xc[tl] = rem - (rem / Do) * Do;
  }

  v8f acc0 = {}, acc1 = {};
  int it = 0;
  for (int t = 0; t < 8; ++t) {
    int a = t >> 2, b = (t >> 1) & 1, c = t & 1;
    const float* ap[2];
#pragma unroll
    for (int tl = 0; tl < 2; ++tl) {
      ap[tl] = in + (size_t)(((2 * zc[tl] + a) * Din + (2 * yc[tl] + b)) * Din +
                             (2 * xc[tl] + c)) * Ci;
    }
    const float* wp = w + (size_t)t * Ci * Co;
    for (int k0 = 0; k0 < Ci; k0 += 32, ++it) {
      __bf16* buf = &sB[it & 1][0];
      stage_B(wp, k0, Ci, Co, col0, buf, tid);
      __syncthreads();
      int ka = k0 + half * 8;
      v16bf B = read_B(buf, lane);
      acc0 = wmma_bf16(load_A_vec(ap[0], ka, rok[0]), B, acc0);
      acc1 = wmma_bf16(load_A_vec(ap[1], ka, rok[1]), B, acc1);
    }
  }
  if (col < Co) {
#pragma unroll
    for (int r = 0; r < 8; ++r) {
      int m = r + half * 8;
      int v0 = vbase + m, v1 = vbase + 16 + m;
      if (v0 < nv) out[(size_t)v0 * os + col] = acc0[r];
      if (v1 < nv) out[(size_t)v1 * os + col] = acc1[r];
    }
  }
}

// 2x2x2 stride-2 transposed conv. grid.z = output parity (8); tiles over
// input voxels so every M-row in a tile shares one weight tap.
__global__ void __launch_bounds__(128) deconv2_wmma(
    const float* __restrict__ in, const float* __restrict__ w,
    float* __restrict__ out, int Din, int Ci, int Co, int os, int nv) {
  __shared__ alignas(32) __bf16 sB[2][512];
  const int p = blockIdx.z;
  const int a = p >> 2, b = (p >> 1) & 1, c = p & 1;
  const int tid  = threadIdx.x;
  const int lane = tid & 31;
  const int wave = tid >> 5;
  const int row  = lane & 15;
  const int half = lane >> 4;
  const int col0 = blockIdx.y << 4;
  const int col  = col0 + row;
  const int vbase = ((blockIdx.x * 4 + wave) * 2) << 4;

  const float* ap[2];
  float rok[2];
#pragma unroll
  for (int tl = 0; tl < 2; ++tl) {
    int vv = vbase + tl * 16 + row;
    rok[tl] = (vv < nv) ? 1.0f : 0.0f;
    if (vv >= nv) vv = nv - 1;
    ap[tl] = in + (size_t)vv * Ci;
  }
  const float* wp = w + (size_t)p * Ci * Co;

  v8f acc0 = {}, acc1 = {};
  int it = 0;
  for (int k0 = 0; k0 < Ci; k0 += 32, ++it) {
    __bf16* buf = &sB[it & 1][0];
    stage_B(wp, k0, Ci, Co, col0, buf, tid);
    __syncthreads();
    int ka = k0 + half * 8;
    v16bf B = read_B(buf, lane);
    acc0 = wmma_bf16(load_A_vec(ap[0], ka, rok[0]), B, acc0);
    acc1 = wmma_bf16(load_A_vec(ap[1], ka, rok[1]), B, acc1);
  }
  if (col < Co) {
    const int HW = Din * Din, Do = Din << 1;
#pragma unroll
    for (int r = 0; r < 8; ++r) {
      int m = r + half * 8;
#pragma unroll
      for (int tl = 0; tl < 2; ++tl) {
        int v2 = vbase + tl * 16 + m;
        if (v2 < nv) {
          int z2 = v2 / HW; int rm = v2 - z2 * HW;
          int y2 = rm / Din; int x2 = rm - y2 * Din;
          size_t oi = (size_t)((2 * z2 + a) * Do + (2 * y2 + b)) * Do + (2 * x2 + c);
          out[oi * os + col] = (tl == 0) ? acc0[r] : acc1[r];
        }
      }
    }
  }
}

// 1x1x1 conv (shortcuts + classifier head).
__global__ void __launch_bounds__(128) conv1_wmma(
    const float* __restrict__ in, const float* __restrict__ w,
    float* __restrict__ out, int Ci, int Co, int os, int nv) {
  __shared__ alignas(32) __bf16 sB[2][512];
  const int tid  = threadIdx.x;
  const int lane = tid & 31;
  const int wave = tid >> 5;
  const int row  = lane & 15;
  const int half = lane >> 4;
  const int col0 = blockIdx.y << 4;
  const int col  = col0 + row;
  const int vbase = ((blockIdx.x * 4 + wave) * 2) << 4;

  const float* ap[2];
  float rok[2];
#pragma unroll
  for (int tl = 0; tl < 2; ++tl) {
    int vv = vbase + tl * 16 + row;
    rok[tl] = (vv < nv) ? 1.0f : 0.0f;
    if (vv >= nv) vv = nv - 1;
    ap[tl] = in + (size_t)vv * Ci;
  }

  v8f acc0 = {}, acc1 = {};
  int it = 0;
  for (int k0 = 0; k0 < Ci; k0 += 32, ++it) {
    __bf16* buf = &sB[it & 1][0];
    stage_B(w, k0, Ci, Co, col0, buf, tid);
    __syncthreads();
    int ka = k0 + half * 8;
    v16bf B = read_B(buf, lane);
    acc0 = wmma_bf16(load_A_vec(ap[0], ka, rok[0]), B, acc0);
    acc1 = wmma_bf16(load_A_vec(ap[1], ka, rok[1]), B, acc1);
  }
  if (col < Co) {
#pragma unroll
    for (int r = 0; r < 8; ++r) {
      int m = r + half * 8;
      int v0 = vbase + m, v1 = vbase + 16 + m;
      if (v0 < nv) out[(size_t)v0 * os + col] = acc0[r];
      if (v1 < nv) out[(size_t)v1 * os + col] = acc1[r];
    }
  }
}

// --------------------------- elementwise / reductions ----------------------

__global__ void zerok(float* p, int n) {
  int i = blockIdx.x * blockDim.x + threadIdx.x;
  if (i < n) p[i] = 0.0f;
}

__global__ void poolmask(const float* __restrict__ m, int Din, float* __restrict__ out) {
  int Do = Din >> 1;
  int n = Do * Do * Do;
  int idx = blockIdx.x * blockDim.x + threadIdx.x;
  if (idx >= n) return;
  int z = idx / (Do * Do); int r = idx - z * Do * Do; int y = r / Do; int x = r - y * Do;
  float mx = 0.0f;
#pragma unroll
  for (int t = 0; t < 8; ++t) {
    int a = t >> 2, b = (t >> 1) & 1, c = t & 1;
    float v = m[((2 * z + a) * Din + (2 * y + b)) * Din + (2 * x + c)];
    mx = fmaxf(mx, v);
  }
  out[idx] = mx;
}

__global__ void maskcount(const float* __restrict__ m, int n, float* out) {
  __shared__ float s[256];
  int tid = threadIdx.x;
  int i = blockIdx.x * 256 + tid;
  s[tid] = (i < n) ? m[i] : 0.0f;
  __syncthreads();
  for (int st = 128; st > 0; st >>= 1) {
    if (tid < st) s[tid] += s[tid + st];
    __syncthreads();
  }
  if (tid == 0) atomicAdd(out, s[0]);
}

// per-channel masked sum / sumsq.  grid = (ceil(nvox/1024), C)
__global__ void bnstats(const float* __restrict__ x, int xs,
                        const float* __restrict__ mask, int nvox,
                        float* __restrict__ stats) {
  const int c = blockIdx.y;
  const int base = blockIdx.x * 1024;
  const int tid = threadIdx.x;
  float s = 0.0f, q = 0.0f;
  for (int i = tid; i < 1024; i += 256) {
    int v = base + i;
    if (v < nvox) {
      float xv = x[(size_t)v * xs + c] * mask[v];
      s += xv; q += xv * xv;
    }
  }
  __shared__ float sh[256], qh[256];
  sh[tid] = s; qh[tid] = q;
  __syncthreads();
  for (int st = 128; st > 0; st >>= 1) {
    if (tid < st) { sh[tid] += sh[tid + st]; qh[tid] += qh[tid + st]; }
    __syncthreads();
  }
  if (tid == 0) {
    atomicAdd(&stats[c], sh[0]);
    atomicAdd(&stats[CMAX + c], qh[0]);
  }
}

// y = ((x-mu)*rsqrt(var+eps)*g + b)*mask  [+ residual] [relu]
__global__ void bnapply(const float* __restrict__ x, int xs,
                        const float* __restrict__ res, int rs,
                        float* __restrict__ out, int os,
                        const float* __restrict__ mask,
                        const float* __restrict__ stats,
                        const float* __restrict__ cnt,
                        const float* __restrict__ g, const float* __restrict__ b,
                        int nvox, int C, int relu) {
  int idx = blockIdx.x * blockDim.x + threadIdx.x;
  if (idx >= nvox * C) return;
  int v = idx / C, c = idx - v * C;
  float n = cnt[0];
  float mu = stats[c] / n;
  float var = stats[CMAX + c] / n - mu * mu;
  float yv = (x[(size_t)v * xs + c] - mu) * rsqrtf(var + 1e-5f) * g[c] + b[c];
  yv *= mask[v];
  if (res) yv += res[(size_t)v * rs + c];
  if (relu) yv = fmaxf(yv, 0.0f);
  out[(size_t)v * os + c] = yv;
}

__global__ void copych(const float* __restrict__ src, int ss,
                       float* __restrict__ dst, int ds, int nvox, int C) {
  int idx = blockIdx.x * blockDim.x + threadIdx.x;
  if (idx >= nvox * C) return;
  int v = idx / C, c = idx - v * C;
  dst[(size_t)v * ds + c] = src[(size_t)v * ss + c];
}

__global__ void biasmask(float* __restrict__ o, const float* __restrict__ bias,
                         const float* __restrict__ mask, int nvox, int C) {
  int idx = blockIdx.x * blockDim.x + threadIdx.x;
  if (idx >= nvox * C) return;
  int v = idx / C, c = idx - v * C;
  o[idx] = (o[idx] + bias[c]) * mask[v];
}

// --------------------------------- host ------------------------------------

extern "C" void kernel_launch(void* const* d_in, const int* in_sizes, int n_in,
                              void* d_out, int out_size, void* d_ws, size_t ws_size,
                              hipStream_t stream) {
  (void)in_sizes; (void)n_in; (void)out_size; (void)ws_size;
  auto P = [&](int i) { return (const float*)d_in[i]; };
  const float* feats = P(0);
  const float* m0 = P(1);

  // ---- workspace bump allocator (f32 elements) ----
  float* ws = (float*)d_ws;
  size_t off = 0;
  auto alloc = [&](size_t nf) { float* p = ws + off; off += nf; return p; };
  float* stats = alloc(2 * CMAX);
  float* cnt   = alloc(8);
  float* m1 = alloc(32768);
  float* m2 = alloc(4096);
  float* m3 = alloc(512);
  float* m4 = alloc(64);
  const int NV0 = 262144, NV1 = 32768, NV2 = 4096, NV3 = 512, NV4 = 64;
  float* x0 = alloc((size_t)NV0 * 32);
  float* x1 = alloc((size_t)NV1 * 32);
  float* x2 = alloc((size_t)NV2 * 64);
  float* x3 = alloc((size_t)NV3 * 128);
  float* x4 = alloc((size_t)NV4 * 256);
  float* tA = alloc((size_t)NV0 * 96);
  float* tB = alloc((size_t)NV0 * 96);
  float* tS = alloc((size_t)NV0 * 96);
  float* t0 = alloc((size_t)NV0 * 96);
  float* cc = alloc((size_t)NV0 * 128);   // concat buffer (reused as y4)
  float* y1 = alloc((size_t)NV3 * 256);
  float* y2 = alloc((size_t)NV2 * 128);
  float* y3 = alloc((size_t)NV1 * 96);
  float* y4 = cc;

  struct Lv { const float* mask; float* cnt; int D; int nv; };
  Lv L0{m0, cnt + 0, 64, NV0};
  Lv L1{m1, cnt + 1, 32, NV1};
  Lv L2{m2, cnt + 2, 16, NV2};
  Lv L3{m3, cnt + 3, 8,  NV3};
  Lv L4{m4, cnt + 4, 4,  NV4};

  // Param leaf index map (JAX pytree: dict keys sorted alphabetically).
  // d_in: 0 feats, 1 mask, then: cls{b,w}=2,3; s1..s4; stem; u1..u4.
  struct RP { const float *w1, *g1, *b1, *w2, *g2, *b2, *wd, *gd, *bd; };
  auto r6 = [&](int i) {  // res w/o downsample: bn1.b,bn1.g,bn2.b,bn2.g,w1,w2
    return RP{P(i + 4), P(i + 1), P(i + 0), P(i + 5), P(i + 3), P(i + 2),
              nullptr, nullptr, nullptr};
  };
  auto r9 = [&](int i) {  // res with wd: bn1.b,g; bn2.b,g; bnd.b,g; w1; w2; wd
    return RP{P(i + 6), P(i + 1), P(i + 0), P(i + 7), P(i + 3), P(i + 2),
              P(i + 8), P(i + 5), P(i + 4)};
  };

  // ---- launch helpers ----
  auto bnrun = [&](const float* x, int xs, const float* res, int rs, float* out,
                   int os, const Lv& L, const float* g, const float* b, int C,
                   int relu) {
    zerok<<<dim3((2 * CMAX + 255) / 256), dim3(256), 0, stream>>>(stats, 2 * CMAX);
    bnstats<<<dim3((L.nv + 1023) / 1024, C), dim3(256), 0, stream>>>(x, xs, L.mask,
                                                                     L.nv, stats);
    int n = L.nv * C;
    bnapply<<<dim3((n + 255) / 256), dim3(256), 0, stream>>>(
        x, xs, res, rs, out, os, L.mask, stats, L.cnt, g, b, L.nv, C, relu);
  };
  auto conv3 = [&](const float* in, int Ci, const float* w, int Co, float* out,
                   const Lv& L) {
    conv3s_wmma<<<dim3((L.nv + 127) / 128, (Co + 15) / 16), dim3(128), 0, stream>>>(
        in, w, out, L.D, Ci, Co, Co, L.nv);
  };
  auto conv1 = [&](const float* in, int Ci, const float* w, int Co, float* out,
                   const Lv& L) {
    conv1_wmma<<<dim3((L.nv + 127) / 128, (Co + 15) / 16), dim3(128), 0, stream>>>(
        in, w, out, Ci, Co, Co, L.nv);
  };
  auto conv2 = [&](const float* in, int Din, int Ci, const float* w, int Co,
                   float* out, const Lv& Lout) {
    conv2d_wmma<<<dim3((Lout.nv + 127) / 128, (Co + 15) / 16), dim3(128), 0,
                  stream>>>(in, w, out, Din, Ci, Co, Co, Lout.nv);
  };
  auto deconv = [&](const float* in, const Lv& Lin, int Ci, const float* w,
                    int Co, float* out) {
    deconv2_wmma<<<dim3((Lin.nv + 127) / 128, (Co + 15) / 16, 8), dim3(128), 0,
                   stream>>>(in, w, out, Lin.D, Ci, Co, Co, Lin.nv);
  };
  // residual block: out = relu(bn2(conv2(relu(bn1(conv1(in))))) + shortcut)
  auto resb = [&](const float* in, int Ci, int Co, const Lv& L, const RP& p,
                  float* u, float* v, float* s, float* out, int os) {
    conv3(in, Ci, p.w1, Co, u, L);
    bnrun(u, Co, nullptr, 0, u, Co, L, p.g1, p.b1, Co, 1);
    conv3(u, Co, p.w2, Co, v, L);
    const float* sc = in; int ss = Ci;
    if (p.wd) {
      conv1(in, Ci, p.wd, Co, s, L);
      bnrun(s, Co, nullptr, 0, s, Co, L, p.gd, p.bd, Co, 0);
      sc = s; ss = Co;
    }
    bnrun(v, Co, sc, ss, out, os, L, p.g2, p.b2, Co, 1);
  };

  // ---- masks + occupied-voxel counts ----
  zerok<<<dim3(1), dim3(256), 0, stream>>>(cnt, 8);
  poolmask<<<dim3((NV1 + 255) / 256), dim3(256), 0, stream>>>(m0, 64, m1);
  poolmask<<<dim3((NV2 + 255) / 256), dim3(256), 0, stream>>>(m1, 32, m2);
  poolmask<<<dim3((NV3 + 255) / 256), dim3(256), 0, stream>>>(m2, 16, m3);
  poolmask<<<dim3(1), dim3(256), 0, stream>>>(m3, 8, m4);
  maskcount<<<dim3(NV0 / 256), dim3(256), 0, stream>>>(m0, NV0, cnt + 0);
  maskcount<<<dim3(NV1 / 256), dim3(256), 0, stream>>>(m1, NV1, cnt + 1);
  maskcount<<<dim3(NV2 / 256), dim3(256), 0, stream>>>(m2, NV2, cnt + 2);
  maskcount<<<dim3((NV3 + 255) / 256), dim3(256), 0, stream>>>(m3, NV3, cnt + 3);
  maskcount<<<dim3(1), dim3(256), 0, stream>>>(m4, NV4, cnt + 4);

  // ---- stem (64^3): 5->32->32 ----
  conv3(feats, 5, P(77), 32, tA, L0);
  bnrun(tA, 32, nullptr, 0, tA, 32, L0, P(74), P(73), 32, 1);
  conv3(tA, 32, P(78), 32, tB, L0);
  bnrun(tB, 32, nullptr, 0, x0, 32, L0, P(76), P(75), 32, 1);

  // ---- encoder stages: down(2^3,s2) + res + res ----
  auto dstage = [&](const float* xin, int ci, int c1, const Lv& Lin, const Lv& Lout,
                    int di, RP p1, RP p2, float* xout) {
    conv2(xin, Lin.D, ci, P(di + 2), ci, tA, Lout);
    bnrun(tA, ci, nullptr, 0, t0, ci, Lout, P(di + 1), P(di + 0), ci, 1);
    resb(t0, ci, c1, Lout, p1, tA, tB, tS, tA, c1);
    resb(tA, c1, c1, Lout, p2, t0, tB, tS, xout, c1);
  };
  dstage(x0, 32,  32,  L0, L1, 4,  r6(7),  r6(13), x1);
  dstage(x1, 32,  64,  L1, L2, 19, r9(22), r6(31), x2);
  dstage(x2, 64,  128, L2, L3, 37, r9(40), r6(49), x3);
  dstage(x3, 128, 256, L3, L4, 55, r9(58), r6(67), x4);

  // ---- decoder stages: deconv + concat + res + res ----
  auto ustage = [&](const float* yin, int ci, const float* skip, int cs,
                    const Lv& Lin, const Lv& Lout, int ui, RP p1, RP p2, int cu,
                    float* yout) {
    int ct = cu + cs;
    deconv(yin, Lin, ci, P(ui + 2), cu, tA);
    bnrun(tA, cu, nullptr, 0, cc, ct, Lout, P(ui + 1), P(ui + 0), cu, 1);
    int n = Lout.nv * cs;
    copych<<<dim3((n + 255) / 256), dim3(256), 0, stream>>>(skip, cs, cc + cu, ct,
                                                            Lout.nv, cs);
    resb(cc, ct, cu, Lout, p1, tA, tB, tS, t0, cu);
    resb(t0, cu, cu, Lout, p2, tA, tB, tS, yout, cu);
  };
  ustage(x4, 256, x3, 128, L4, L3, 94,  r9(79),  r6(88),  256, y1);
  ustage(y1, 256, x2, 64,  L3, L2, 112, r9(97),  r6(106), 128, y2);
  ustage(y2, 128, x1, 32,  L2, L1, 130, r9(115), r6(124), 96,  y3);
  ustage(y3, 96,  x0, 32,  L1, L0, 148, r9(133), r6(142), 96,  y4);

  // ---- classifier head: 1x1 conv 96->20, + bias, * mask ----
  conv1(y4, 96, P(3), 20, (float*)d_out, L0);
  int nout = NV0 * 20;
  biasmask<<<dim3((nout + 255) / 256), dim3(256), 0, stream>>>(
      (float*)d_out, P(2), m0, NV0, 20);
}
